// MetNoulliLatentModel_36687610643007
// MI455X (gfx1250) — matrix operand order repaired and checked by reading, not compiled, not run on hardware
//
#include <hip/hip_runtime.h>

// Problem dims (fixed by the reference)
#define NN 10000
#define DD 128
#define TTOT 128
#define KK 32
#define PP 64

// Tiling
#define N_TILE 16
#define T_TILE 32
#define D_TILE 16

// LDS strides (in 16-bit element units unless noted)
#define THETA_NSTRIDE 40                        // >=32, mult of 8 (16B align), bank-spread
#define THETA_TSTRIDE (N_TILE * THETA_NSTRIDE)  // 640
#define PHI_DSTRIDE   40
#define PHI_TSTRIDE   (D_TILE * PHI_DSTRIDE + 8) // 648 (16B-aligned stride)
#define OUT_TSTRIDE   33                        // float units, bank padding

typedef __attribute__((ext_vector_type(16))) __bf16        v16bf;
typedef __attribute__((ext_vector_type(8)))  float         v8f;
typedef __attribute__((ext_vector_type(4)))  unsigned int  u32x4;

union FragBF { v16bf v; u32x4 q[2]; unsigned short s[16]; };

__device__ __forceinline__ unsigned short f2bf(float f) {
  union { float f; unsigned u; } x; x.f = f;
  unsigned r = x.u + 0x7FFFu + ((x.u >> 16) & 1u);   // round-to-nearest-even
  return (unsigned short)(r >> 16);
}

__global__ __launch_bounds__(256)
void metnoulli_fused(const float* __restrict__ cov,
                     const float* __restrict__ gamma,
                     const float* __restrict__ lam,
                     const float* __restrict__ phi,
                     const float* __restrict__ psi,
                     const float* __restrict__ base,
                     float* __restrict__ probs,
                     float* __restrict__ theta)
{
  __shared__ alignas(16) unsigned short thetaBf[T_TILE * THETA_TSTRIDE]; // [t][n][k] bf16
  __shared__ alignas(16) unsigned short phiBf  [T_TILE * PHI_TSTRIDE];   // [t][d][k] bf16
  __shared__ alignas(16) float outStage[N_TILE * D_TILE * OUT_TSTRIDE];  // [n][d][t]
  __shared__ alignas(16) float covEff[N_TILE * KK];                      // [n][k]

  const int tid  = threadIdx.x;
  const int lane = tid & 31;
  const int wave = tid >> 5;
  const int n0 = blockIdx.x * N_TILE;
  const int t0 = blockIdx.y * T_TILE;

  // ---- 1. cov_effect = covariates @ gamma, tile (16 x 32) ----
  for (int idx = tid; idx < N_TILE * KK; idx += 256) {
    const int n = idx >> 5, k = idx & 31;
    const float* crow = cov + (long)(n0 + n) * PP;
    float acc = 0.f;
#pragma unroll
    for (int p = 0; p < PP; ++p) acc = __fmaf_rn(crow[p], gamma[p * KK + k], acc);
    covEff[idx] = acc;
  }
  __syncthreads();

  // ---- 2. softmax over K per (n,t); lane = t so every lambda read is coalesced ----
#pragma unroll
  for (int rep = 0; rep < 2; ++rep) {
    const int n = wave + rep * 8;
    const float* lrow = lam + (long)(n0 + n) * KK * TTOT + (t0 + lane);
    float v[KK];
    float mx = -3.4e38f;
#pragma unroll
    for (int k = 0; k < KK; ++k) {
      const float x = lrow[(long)k * TTOT] + covEff[n * KK + k];
      v[k] = x;
      mx = fmaxf(mx, x);
    }
    float s = 0.f;
#pragma unroll
    for (int k = 0; k < KK; ++k) { v[k] = __expf(v[k] - mx); s += v[k]; }
    const float inv = __frcp_rn(s);
    float* trow = theta + (long)(n0 + n) * KK * TTOT + (t0 + lane);
    unsigned short* tb = thetaBf + lane * THETA_TSTRIDE + n * THETA_NSTRIDE;
#pragma unroll
    for (int k = 0; k < KK; ++k) {
      const float th = v[k] * inv;
      trow[(long)k * TTOT] = th;   // coalesced theta output
      tb[k] = f2bf(th);            // bf16 copy for WMMA A
    }
  }
  __syncthreads();

  // ---- 3. hoist WMMA A fragments for this wave's 4 t-values ----
  // A 16-bit 16x32 layout: lane M = lane%16; VGPR0-3: K=8*half+0..7; VGPR4-7: K=16+8*half+0..7
  const int m    = lane & 15;
  const int half = lane >> 4;
  FragBF afrag[4];
#pragma unroll
  for (int i = 0; i < 4; ++i) {
    const int tl = wave + 8 * i;
    const unsigned short* p = thetaBf + tl * THETA_TSTRIDE + m * THETA_NSTRIDE + 8 * half;
    afrag[i].q[0] = *(const u32x4*)(p);
    afrag[i].q[1] = *(const u32x4*)(p + 16);
  }

  // ---- 4. loop over 8 d-tiles of 16 ----
  for (int dt = 0; dt < DD / D_TILE; ++dt) {
    const int d0 = dt * D_TILE;

    // stage phi_logits = phi + psi as bf16, layout [t][d][k] (k contiguous); coalesced reads
#pragma unroll 4
    for (int j = 0; j < 64; ++j) {
      const int r = wave + 8 * j;             // r = k*16 + d, 512 rows
      const int k = r >> 4, d = r & 15;
      const float x = phi[((long)k * DD + d0 + d) * TTOT + t0 + lane] + psi[k * DD + d0 + d];
      phiBf[lane * PHI_TSTRIDE + d * PHI_DSTRIDE + k] = f2bf(x);
    }
    __syncthreads();

    // one v_wmma_f32_16x16x32_bf16 per (t, d-tile); K=32 in a single instruction
    // B 16-bit 32x16 layout: lane N = lane%16; VGPR0-3: K=8*half+0..7; VGPR4-7: K=16+8*half+0..7
#pragma unroll
    for (int i = 0; i < 4; ++i) {
      const int tl = wave + 8 * i;
      FragBF bfrag;
      const unsigned short* p = phiBf + tl * PHI_TSTRIDE + m * PHI_DSTRIDE + 8 * half;
      bfrag.q[0] = *(const u32x4*)(p);
      bfrag.q[1] = *(const u32x4*)(p + 16);

      v8f c = {0.f, 0.f, 0.f, 0.f, 0.f, 0.f, 0.f, 0.f};
      c = __builtin_amdgcn_wmma_f32_16x16x32_bf16(false, afrag[i].v,
                                                  false, bfrag.v,
                                                  (short)0, c, false, false);

      const float bl = base[(d0 + m) * TTOT + t0 + tl];
      // C/D layout: VGPR r -> M = r + 8*half, N = lane%16
      float* op = outStage + (8 * half * D_TILE + m) * OUT_TSTRIDE + tl;
#pragma unroll
      for (int r = 0; r < 8; ++r) {
        const float logit = c[r] + bl;
        op[r * (D_TILE * OUT_TSTRIDE)] = __frcp_rn(1.f + __expf(-logit));
      }
    }
    __syncthreads();

    // coalesced writeout: 256 (n,d) rows x 32 contiguous t floats (lane = t)
#pragma unroll 4
    for (int j = 0; j < 32; ++j) {
      const int r = wave + 8 * j;
      const int n = r >> 4, d = r & 15;
      probs[((long)(n0 + n) * DD + d0 + d) * TTOT + t0 + lane] =
          outStage[(n * D_TILE + d) * OUT_TSTRIDE + lane];
    }
    __syncthreads();  // guard phiBf/outStage reuse next iteration
  }
}

extern "C" void kernel_launch(void* const* d_in, const int* in_sizes, int n_in,
                              void* d_out, int out_size, void* d_ws, size_t ws_size,
                              hipStream_t stream) {
  const float* cov   = (const float*)d_in[0];  // (N,P)
  const float* gamma = (const float*)d_in[1];  // (P,K)
  const float* lam   = (const float*)d_in[2];  // (N,K,T)
  const float* phi   = (const float*)d_in[3];  // (K,D,T)
  const float* psi   = (const float*)d_in[4];  // (K,D)
  const float* base  = (const float*)d_in[5];  // (D,T)

  float* probs = (float*)d_out;                          // (N,D,T)
  float* theta = probs + (size_t)NN * DD * TTOT;         // (N,K,T)

  dim3 grid(NN / N_TILE, TTOT / T_TILE);                 // 625 x 4
  metnoulli_fused<<<grid, 256, 0, stream>>>(cov, gamma, lam, phi, psi, base, probs, theta);
}
